// DifferentiableMemory_16853451669706
// MI455X (gfx1250) — matrix operand order repaired and checked by reading, not compiled, not run on hardware
//
#include <hip/hip_runtime.h>
#include <hip/hip_bf16.h>

// ---------------------------------------------------------------------------
// Problem constants (match reference)
// ---------------------------------------------------------------------------
constexpr int BQ = 2048;     // queries
constexpr int H  = 512;      // hidden
constexpr int M  = 131072;   // memories
constexpr int TOPK = 32;

// Tiling for the WMMA similarity kernel
constexpr int TB   = 32;     // query rows per block (2 row tiles of 16)
constexpr int MT   = 256;    // memory columns per chunk (16 col tiles of 16)
constexpr int QCAP = 256;    // per-row candidate queue capacity (== MT, cannot overflow)
constexpr float NEGV = -1.0e9f;

// B slab staging (async path): per wave, 32 cols x 32 K values (bf16)
constexpr int COL_STRIDE = 40;              // elements (80 B) -> bank-conflict-free ds_load_b128
constexpr int SLAB_ELEMS = 32 * COL_STRIDE; // per-wave slab size in elements

typedef __attribute__((ext_vector_type(16))) __bf16 v16bf;
typedef __attribute__((ext_vector_type(8)))  __bf16 v8bf;
typedef __attribute__((ext_vector_type(8)))  float  v8f;
typedef __attribute__((ext_vector_type(4)))  int    v4i;

// Async global->LDS copy path (gfx1250). Guarded: falls back to direct global
// loads if the builtin is not declared by this toolchain.
#if defined(__gfx1250__) && __has_builtin(__builtin_amdgcn_global_load_async_to_lds_b128)
#define HAVE_ASYNC_LDS 1
typedef v4i __attribute__((address_space(1))) gv4i_t;   // global int4*
typedef v4i __attribute__((address_space(3))) lv4i_t;   // LDS int4*
#if __has_builtin(__builtin_amdgcn_s_wait_asynccnt)
#define WAIT_ASYNC(n) __builtin_amdgcn_s_wait_asynccnt(n)
#else
#define WAIT_ASYNC(n) asm volatile("s_wait_asynccnt %0" :: "i"(n) : "memory")
#endif
#else
#define HAVE_ASYNC_LDS 0
#endif

// ws layout (bytes)
constexpr size_t WS_KEYS_BF = 0;                                  // M*H*2   = 134217728
constexpr size_t WS_QN_BF   = WS_KEYS_BF + (size_t)M * H * 2;     // BQ*H*2  = 2097152
constexpr size_t WS_QN_F32  = WS_QN_BF   + (size_t)BQ * H * 2;    // BQ*H*4  = 4194304
constexpr size_t WS_TOPV    = WS_QN_F32  + (size_t)BQ * H * 4;
constexpr size_t WS_TOPI    = WS_TOPV    + (size_t)BQ * TOPK * 4;

// out layout (floats)
constexpr size_t OUT_KEYS = 0;
constexpr size_t OUT_VALS = (size_t)BQ * TOPK * H;
constexpr size_t OUT_WTS  = 2 * (size_t)BQ * TOPK * H;

// ---------------------------------------------------------------------------
// Kernel 1: normalize queries -> fp32 + bf16 copies
// ---------------------------------------------------------------------------
__global__ __launch_bounds__(128) void prep_q_kernel(
    const float* __restrict__ q, float* __restrict__ qn_f, __bf16* __restrict__ qn_bf)
{
    __shared__ float red[128];
    const int b = blockIdx.x, tid = threadIdx.x;
    const float* row = q + (size_t)b * H;
    float ss = 0.f;
    for (int i = tid; i < H; i += 128) { float x = row[i]; ss += x * x; }
    red[tid] = ss;
    __syncthreads();
    for (int s = 64; s > 0; s >>= 1) {
        if (tid < s) red[tid] += red[tid + s];
        __syncthreads();
    }
    const float inv = 1.0f / (sqrtf(red[0]) + 1e-12f);
    for (int i = tid; i < H; i += 128) {
        float x = row[i] * inv;
        qn_f[(size_t)b * H + i] = x;
        qn_bf[(size_t)b * H + i] = (__bf16)x;
    }
}

// ---------------------------------------------------------------------------
// Kernel 2: memory_keys f32 -> bf16 (128 MB result fits in the 192 MB L2)
// ---------------------------------------------------------------------------
__global__ __launch_bounds__(256) void conv_keys_kernel(
    const float* __restrict__ src, __bf16* __restrict__ dst, long n)
{
    long i = ((long)blockIdx.x * blockDim.x + threadIdx.x) * 8;
    const long stride = (long)gridDim.x * blockDim.x * 8;
    for (; i < n; i += stride) {
        float4 a = *(const float4*)(src + i);
        float4 b = *(const float4*)(src + i + 4);
        v8bf o;
        o[0] = (__bf16)a.x; o[1] = (__bf16)a.y; o[2] = (__bf16)a.z; o[3] = (__bf16)a.w;
        o[4] = (__bf16)b.x; o[5] = (__bf16)b.y; o[6] = (__bf16)b.z; o[7] = (__bf16)b.w;
        *(v8bf*)(dst + i) = o;
    }
}

// ---------------------------------------------------------------------------
// Kernel 3: fused WMMA similarity + masking + streaming top-32
//   grid = BQ/TB blocks, 512 threads (16 waves).
//   Wave w: row tile (w&1), column tiles {2*(w>>1), 2*(w>>1)+1} of each chunk.
//   A row tile (16 x 512 bf16) lives in 128 VGPRs for the whole kernel.
//   B is staged per-wave into double-buffered LDS slabs with async copies
//   (ASYNCcnt-pipelined one k-slab ahead, incl. across chunk boundaries).
// ---------------------------------------------------------------------------
__global__ __launch_bounds__(512) void sim_topk_kernel(
    const __bf16* __restrict__ qn_bf, const __bf16* __restrict__ keys_bf,
    const int* __restrict__ uid, const int* __restrict__ slen,
    const int* __restrict__ muid, const int* __restrict__ mslen,
    const float* __restrict__ temp,
    float* __restrict__ out_topv, int* __restrict__ out_topi)
{
#if HAVE_ASYNC_LDS
    __shared__ __align__(16) __bf16 sB[2][16][SLAB_ELEMS];   // 80 KB B slabs
#endif
    __shared__ float sQv[TB * QCAP];                          // 32 KB candidate values
    __shared__ unsigned short sQi[TB * QCAP];                 // 16 KB in-chunk col offsets
    __shared__ float sTopV[TB][TOPK];                         // sorted ascending; [0] = min
    __shared__ int   sTopI[TB][TOPK];
    __shared__ float sThr[TB];
    __shared__ int   sQn[TB];
    __shared__ int   sUid[TB], sSlen[TB];
    __shared__ int   sMuid[MT], sMslen[MT];

    const int tid   = threadIdx.x;
    const int wave  = tid >> 5;
    const int lane  = tid & 31;
    const int l16   = lane & 15;
    const int khalf = lane >> 4;
    const int rtile = wave & 1;            // which 16-row tile of the block
    const int cpair = wave >> 1;           // which pair of 16-col tiles
    const int wbase = cpair * 32;          // first column (within chunk) of this wave
    const int b0    = blockIdx.x * TB;
    const float invT = 1.0f / temp[0];

    // ---- A row tile resident in registers: 16 fragments x 8 VGPRs ----
    v16bf Afrag[16];
    {
        const __bf16* arow = qn_bf + (size_t)(b0 + rtile * 16 + l16) * H;
        #pragma unroll
        for (int k = 0; k < 16; ++k) {
            v8bf lo = *(const v8bf*)&arow[k * 32 + khalf * 16];
            v8bf hi = *(const v8bf*)&arow[k * 32 + khalf * 16 + 8];
            Afrag[k] = __builtin_shufflevector(lo, hi,
                0,1,2,3,4,5,6,7,8,9,10,11,12,13,14,15);
        }
    }

    if (tid < TB) {
        sUid[tid] = uid[b0 + tid];
        sSlen[tid] = slen[b0 + tid];
        sThr[tid] = -3.0e38f;
        sQn[tid] = 0;
        for (int j = 0; j < TOPK; ++j) { sTopV[tid][j] = NEGV; sTopI[tid][j] = j; }
    }
    __syncthreads();

#if HAVE_ASYNC_LDS
    // Issue one k-slab of B (32 cols x 32 K, 64B/col) into this wave's region.
    // 128 chunks of 16B, 4 per lane, tracked with ASYNCcnt.
    // Builtin signature (probe-confirmed arg0): (v4i AS1* src, v4i AS3* dst, imm, imm).
    #define ISSUE_SLAB(bufi, m0_, k0_) do {                                     \
        const __bf16* gbase_ = keys_bf + (size_t)((m0_) + wbase) * H + (k0_);   \
        __bf16* lbase_ = &sB[(bufi)][wave][0];                                  \
        _Pragma("unroll")                                                       \
        for (int j_ = 0; j_ < 4; ++j_) {                                        \
            const int q_ = lane + 32 * j_;                                      \
            const int cc_ = q_ >> 2, part_ = q_ & 3;                            \
            gv4i_t* gp_ = (gv4i_t*)(unsigned long long)                         \
                (gbase_ + (size_t)cc_ * H + part_ * 8);                         \
            lv4i_t* lp_ = (lv4i_t*)(unsigned)(unsigned long long)               \
                (lbase_ + cc_ * COL_STRIDE + part_ * 8);                        \
            __builtin_amdgcn_global_load_async_to_lds_b128(gp_, lp_, 0, 0);     \
        }                                                                       \
    } while (0)

    ISSUE_SLAB(0, 0, 0);   // prologue: slab for chunk 0, k-step 0
#endif

    for (int m0 = 0; m0 < M; m0 += MT) {
        if (tid < MT) { sMuid[tid] = muid[m0 + tid]; sMslen[tid] = mslen[m0 + tid]; }
        __syncthreads();

        v8f acc0 = {};   // col tile c = cpair*2
        v8f acc1 = {};   // col tile c = cpair*2 + 1
#if !HAVE_ASYNC_LDS
        const __bf16* bcol0 = keys_bf + (size_t)(m0 + wbase + l16) * H;
        const __bf16* bcol1 = bcol0 + (size_t)16 * H;
#endif
        #pragma unroll
        for (int i = 0; i < 16; ++i) {
            v8bf b0lo, b0hi, b1lo, b1hi;
#if HAVE_ASYNC_LDS
            // keep exactly one slab (4 async ops) in flight while computing
            if (i < 15) {
                ISSUE_SLAB((i + 1) & 1, m0, (i + 1) * 32);
                WAIT_ASYNC(4);
            } else if (m0 + MT < M) {
                ISSUE_SLAB(0, m0 + MT, 0);     // warm next chunk's first slab
                WAIT_ASYNC(4);
            } else {
                WAIT_ASYNC(0);
            }
            const __bf16* slab = &sB[i & 1][wave][0];
            b0lo = *(const v8bf*)&slab[(l16)      * COL_STRIDE + khalf * 16];
            b0hi = *(const v8bf*)&slab[(l16)      * COL_STRIDE + khalf * 16 + 8];
            b1lo = *(const v8bf*)&slab[(16 + l16) * COL_STRIDE + khalf * 16];
            b1hi = *(const v8bf*)&slab[(16 + l16) * COL_STRIDE + khalf * 16 + 8];
#else
            b0lo = *(const v8bf*)&bcol0[i * 32 + khalf * 16];
            b0hi = *(const v8bf*)&bcol0[i * 32 + khalf * 16 + 8];
            b1lo = *(const v8bf*)&bcol1[i * 32 + khalf * 16];
            b1hi = *(const v8bf*)&bcol1[i * 32 + khalf * 16 + 8];
#endif
            v16bf bb0 = __builtin_shufflevector(b0lo, b0hi,
                0,1,2,3,4,5,6,7,8,9,10,11,12,13,14,15);
            v16bf bb1 = __builtin_shufflevector(b1lo, b1hi,
                0,1,2,3,4,5,6,7,8,9,10,11,12,13,14,15);
            acc0 = __builtin_amdgcn_wmma_f32_16x16x32_bf16(
                       false, Afrag[i], false, bb0, (short)0, acc0, false, false);
            acc1 = __builtin_amdgcn_wmma_f32_16x16x32_bf16(
                       false, Afrag[i], false, bb1, (short)0, acc1, false, false);
        }

        // ---- filter: lane owns cols (wbase+l16, wbase+l16+16), 8 rows each ----
        union F8 { v8f v; float f[8]; } u0, u1;
        u0.v = acc0; u1.v = acc1;
        const int co0 = wbase + l16;
        const int co1 = co0 + 16;
        const int uid0 = sMuid[co0], sl0 = sMslen[co0];
        const int uid1 = sMuid[co1], sl1 = sMslen[co1];
        #pragma unroll
        for (int v = 0; v < 8; ++v) {
            const int row = rtile * 16 + khalf * 8 + v;
            const int ruid = sUid[row], rsl = sSlen[row];
            const float thr = sThr[row];
            const float val0 = u0.f[v] * invT;
            const float val1 = u1.f[v] * invT;
            if (!((uid0 == ruid) && (sl0 >= rsl)) && val0 > thr) {
                int pos = atomicAdd(&sQn[row], 1);
                sQv[row * QCAP + pos] = val0;
                sQi[row * QCAP + pos] = (unsigned short)co0;
            }
            if (!((uid1 == ruid) && (sl1 >= rsl)) && val1 > thr) {
                int pos = atomicAdd(&sQn[row], 1);
                sQv[row * QCAP + pos] = val1;
                sQi[row * QCAP + pos] = (unsigned short)co1;
            }
        }
        __syncthreads();

        // ---- owner phase: thread t merges row t's candidates ----
        if (tid < TB) {
            int n = sQn[tid]; if (n > QCAP) n = QCAP;
            float* tv = sTopV[tid];
            int*   ti = sTopI[tid];
            for (int j = 0; j < n; ++j) {
                const float v = sQv[tid * QCAP + j];
                const int  ix = m0 + (int)sQi[tid * QCAP + j];
                if (v > tv[0]) {
                    int p = 1;
                    while (p < TOPK && tv[p] < v) { tv[p-1] = tv[p]; ti[p-1] = ti[p]; ++p; }
                    tv[p-1] = v; ti[p-1] = ix;
                }
            }
            sThr[tid] = tv[0];
            sQn[tid] = 0;
        }
        __syncthreads();
    }

    if (tid < TB) {
        for (int j = 0; j < TOPK; ++j) {
            out_topv[(size_t)(b0 + tid) * TOPK + j] = sTopV[tid][j];
            out_topi[(size_t)(b0 + tid) * TOPK + j] = sTopI[tid][j];
        }
    }
}

// ---------------------------------------------------------------------------
// Kernel 4: exact fp32 re-score of the 32 picks, sort desc, softmax, gather
// ---------------------------------------------------------------------------
__global__ __launch_bounds__(256) void finalize_kernel(
    const float* __restrict__ qn_f, const float* __restrict__ keys_f,
    const float* __restrict__ vals_f, const int* __restrict__ topi,
    const float* __restrict__ temp, float* __restrict__ out)
{
    __shared__ int   sIdx[TOPK];
    __shared__ float sPart[TOPK][8];
    __shared__ float sSim[TOPK];
    __shared__ float sOrdV[TOPK];
    __shared__ int   sOrdI[TOPK];

    const int b = blockIdx.x, tid = threadIdx.x;
    if (tid < TOPK) sIdx[tid] = topi[(size_t)b * TOPK + tid];
    __syncthreads();

    // exact fp32 dot products: 8 threads per selected row
    {
        const int j = tid >> 3, seg = tid & 7;
        const float* q = qn_f + (size_t)b * H;
        const float* k = keys_f + (size_t)sIdx[j] * H;
        float s = 0.f;
        for (int i = seg * 64; i < seg * 64 + 64; ++i) s += q[i] * k[i];
        sPart[j][seg] = s;
    }
    __syncthreads();
    if (tid < TOPK) {
        float s = 0.f;
        #pragma unroll
        for (int e = 0; e < 8; ++e) s += sPart[tid][e];
        sSim[tid] = s / temp[0];
    }
    __syncthreads();

    if (tid == 0) {
        for (int j = 0; j < TOPK; ++j) { sOrdV[j] = sSim[j]; sOrdI[j] = sIdx[j]; }
        for (int a = 1; a < TOPK; ++a) {          // insertion sort, descending
            float v = sOrdV[a]; int ix = sOrdI[a]; int p = a - 1;
            while (p >= 0 && sOrdV[p] < v) { sOrdV[p+1] = sOrdV[p]; sOrdI[p+1] = sOrdI[p]; --p; }
            sOrdV[p+1] = v; sOrdI[p+1] = ix;
        }
        const float mx = sOrdV[0];
        float sum = 0.f;
        float w[TOPK];
        for (int j = 0; j < TOPK; ++j) { w[j] = __expf(sOrdV[j] - mx); sum += w[j]; }
        const float inv = 1.0f / sum;
        float* wout = out + OUT_WTS + (size_t)b * TOPK;
        for (int j = 0; j < TOPK; ++j) wout[j] = w[j] * inv;
    }
    __syncthreads();

    // gather the selected key/value rows (float4 vectorized)
    float* ok = out + OUT_KEYS + (size_t)b * TOPK * H;
    float* ov = out + OUT_VALS + (size_t)b * TOPK * H;
    for (int e = tid; e < TOPK * (H / 4); e += 256) {
        const int j = e >> 7, q4 = e & 127;
        const float4* sk = (const float4*)(keys_f + (size_t)sOrdI[j] * H);
        const float4* sv = (const float4*)(vals_f + (size_t)sOrdI[j] * H);
        ((float4*)(ok + j * H))[q4] = sk[q4];
        ((float4*)(ov + j * H))[q4] = sv[q4];
    }
}

// ---------------------------------------------------------------------------
extern "C" void kernel_launch(void* const* d_in, const int* in_sizes, int n_in,
                              void* d_out, int out_size, void* d_ws, size_t ws_size,
                              hipStream_t stream)
{
    const float* q     = (const float*)d_in[0];
    const int*   uid   = (const int*)d_in[1];
    const int*   slen  = (const int*)d_in[2];
    const float* mkeys = (const float*)d_in[3];
    const float* mvals = (const float*)d_in[4];
    const int*   muid  = (const int*)d_in[5];
    const int*   mslen = (const int*)d_in[6];
    const float* temp  = (const float*)d_in[7];
    float* out = (float*)d_out;

    char* ws = (char*)d_ws;
    __bf16* keys_bf = (__bf16*)(ws + WS_KEYS_BF);
    __bf16* qn_bf   = (__bf16*)(ws + WS_QN_BF);
    float*  qn_f    = (float*)(ws + WS_QN_F32);
    float*  topv    = (float*)(ws + WS_TOPV);
    int*    topi    = (int*)(ws + WS_TOPI);

    prep_q_kernel<<<BQ, 128, 0, stream>>>(q, qn_f, qn_bf);

    const long nk = (long)M * H;
    conv_keys_kernel<<<(int)(nk / (8 * 256)), 256, 0, stream>>>(mkeys, keys_bf, nk);

    sim_topk_kernel<<<BQ / TB, 512, 0, stream>>>(
        qn_bf, keys_bf, uid, slen, muid, mslen, temp, topv, topi);

    finalize_kernel<<<BQ, 256, 0, stream>>>(qn_f, mkeys, mvals, topi, temp, out);
}
